// GraphAttention_44006234915656
// MI455X (gfx1250) — matrix operand order repaired and checked by reading, not compiled, not run on hardware
//
#include <hip/hip_runtime.h>

typedef __attribute__((ext_vector_type(16))) _Float16 v16h;
typedef __attribute__((ext_vector_type(8)))  float    v8f;

#define NBATCH 4096
#define Mn     64
#define Hd     128
#define NH     8
#define QROWS  80    // 65 rows padded to 5 tiles of 16
#define QS     132   // LDS row stride (floats), avoids 64-bank alignment

__global__ __launch_bounds__(256) void graph_attn_kernel(
    const float* __restrict__ value,   // [B,1,H]
    const float* __restrict__ query,   // [B,M,H]
    const float* __restrict__ Wq,      // [NH,H,H]
    const float* __restrict__ bq,      // [NH,H]
    const float* __restrict__ Wo,      // [NH*H,H]
    const float* __restrict__ bo,      // [H]
    const int*   __restrict__ nn_p,    // scalar num_neighbors
    float* __restrict__ out)           // [B,M,H]
{
  __shared__ float qcur[QROWS * QS];
  __shared__ float vvec[Hd];
  __shared__ float eLds[72];
  __shared__ float wLds[72];
  __shared__ float ctxLds[2 * NH * Hd];
  __shared__ __attribute__((aligned(16))) float outLds[2 * Hd];

  const int t    = threadIdx.x;
  const int b    = blockIdx.x;
  const int lane = t & 31;
  const int w    = t >> 5;

  // ---------- load q = concat(value[b], query[b]) into LDS, zero-pad ----------
  const float* vb = value + (size_t)b * Hd;
  const float* qb = query + (size_t)b * Mn * Hd;
  for (int idx = t; idx < 65 * Hd; idx += 256) {
    int n = idx >> 7, h = idx & (Hd - 1);
    qcur[n * QS + h] = (n == 0) ? vb[h] : qb[(n - 1) * Hd + h];
  }
  for (int idx = t; idx < (QROWS - 65) * Hd; idx += 256) {
    qcur[(65 + (idx >> 7)) * QS + (idx & (Hd - 1))] = 0.f;
  }
  if (t < Hd) vvec[t] = vb[t];
  __syncthreads();

  // WMMA operand addressing (wave32 layouts, 16-bit A 16x32 / B 32x16)
  const int lrow = lane & 15;
  const int col  = w * 16 + lrow;       // this wave owns q_new columns [16w, 16w+16)
  const int ka   = (lane >> 4) * 16;    // A: lanes 0-15 K{0-15},      16-31 K{16-31}
  const int kb   = (lane >> 4) * 16;    // B: lanes 0-15 K{0-15},      16-31 K{16-31}

  for (int head = 0; head < NH; ++head) {
    const float* Wqi = Wq + (size_t)head * Hd * Hd;

    // -------- q_new = f16(q) @ f16(Wq[head]) + bq[head], f32 accumulate --------
    v8f acc[5];
    #pragma unroll
    for (int m = 0; m < 5; ++m)
      #pragma unroll
      for (int e = 0; e < 8; ++e) acc[m][e] = 0.f;

    #pragma unroll
    for (int kk = 0; kk < 4; ++kk) {
      const int k0a = kk * 32 + ka;
      const int k0b = kk * 32 + kb;
      v16h bv;
      #pragma unroll
      for (int j = 0; j < 16; ++j)
        bv[j] = (_Float16)Wqi[(k0b + j) * Hd + col];
      #pragma unroll
      for (int m = 0; m < 5; ++m) {
        const float* ar = &qcur[(m * 16 + lrow) * QS];
        v16h av;
        #pragma unroll
        for (int j = 0; j < 8; ++j) {
          av[j]     = (_Float16)ar[k0a + j];
          av[j + 8] = (_Float16)ar[k0a + 8 + j];
        }
        acc[m] = __builtin_amdgcn_wmma_f32_16x16x32_f16(
            false, av, false, bv, (short)0, acc[m], false, false);
      }
    }
    __syncthreads();  // all reads of qcur done -> safe to overwrite in place

    const float bias = bq[head * Hd + col];
    const int   rofs = (lane >> 4) * 8;   // D: vgpr e -> row e (lanes 0-15) / e+8 (16-31)
    #pragma unroll
    for (int m = 0; m < 5; ++m)
      #pragma unroll
      for (int e = 0; e < 8; ++e)
        qcur[(m * 16 + e + rofs) * QS + col] = acc[m][e] + bias;
    __syncthreads();

    // -------- e[n] = value . q[n], n = 0..64 --------
    if (t < 65) {
      const float* qr = &qcur[t * QS];
      float s = 0.f;
      #pragma unroll 4
      for (int h = 0; h < Hd; ++h) s += vvec[h] * qr[h];
      eLds[t] = s;
    }
    __syncthreads();

    // -------- swish + softmax over 65 entries (wave 0) --------
    if (t < 32) {
      float e0 = eLds[lane], e1 = eLds[lane + 32];
      float s0 = e0 / (1.f + __expf(-e0));
      float s1 = e1 / (1.f + __expf(-e1));
      float s2 = -1e30f;
      if (lane == 0) { float e2 = eLds[64]; s2 = e2 / (1.f + __expf(-e2)); }
      float mx = fmaxf(fmaxf(s0, s1), s2);
      #pragma unroll
      for (int o = 16; o >= 1; o >>= 1) mx = fmaxf(mx, __shfl_xor(mx, o, 32));
      float x0 = __expf(s0 - mx), x1 = __expf(s1 - mx);
      float x2 = (lane == 0) ? __expf(s2 - mx) : 0.f;
      float sm = x0 + x1 + x2;
      #pragma unroll
      for (int o = 16; o >= 1; o >>= 1) sm += __shfl_xor(sm, o, 32);
      float inv = 1.f / sm;
      wLds[lane]      = x0 * inv;
      wLds[lane + 32] = x1 * inv;
      if (lane == 0) wLds[64] = x2 * inv;
    }
    __syncthreads();

    // -------- two distinct context rows: softmax-weighted & uniform mean --------
    {
      const int h     = t & (Hd - 1);
      const int which = t >> 7;              // 0: masked-in rows, 1: masked-out rows
      float s = 0.f;
      if (which == 0) {
        for (int n = 0; n < 65; ++n) s += wLds[n] * qcur[n * QS + h];
      } else {
        for (int n = 0; n < 65; ++n) s += qcur[n * QS + h];
        s *= (1.f / 65.f);
      }
      ctxLds[which * (NH * Hd) + head * Hd + h] = s;
    }
    __syncthreads();
  }

  // -------- x = concat(contexts) @ Wo + bo, relu (only 2 distinct rows) --------
  {
    const int c = t & (Hd - 1);
    const int r = t >> 7;
    const float* ctx = &ctxLds[r * (NH * Hd)];
    float s = bo[c];
    for (int j = 0; j < NH * Hd; ++j) s += ctx[j] * Wo[(size_t)j * Hd + c];
    outLds[r * Hd + c] = fmaxf(s, 0.f);
  }
  __syncthreads();

  // -------- broadcast the two rows across m, coalesced float4 stores --------
  {
    const int cutoff = *nn_p + 1;           // rows m < cutoff use softmax row
    const int c4 = (t & 31) << 2;
    const int mg = t >> 5;
    float4 vA = *(const float4*)&outLds[c4];
    float4 vB = *(const float4*)&outLds[Hd + c4];
    float* ob = out + (size_t)b * Mn * Hd;
    #pragma unroll
    for (int mm = 0; mm < 8; ++mm) {
      int m = mg * 8 + mm;
      float4 vv = (m < cutoff) ? vA : vB;
      *(float4*)&ob[m * Hd + c4] = vv;
    }
  }
}

extern "C" void kernel_launch(void* const* d_in, const int* in_sizes, int n_in,
                              void* d_out, int out_size, void* d_ws, size_t ws_size,
                              hipStream_t stream) {
  (void)in_sizes; (void)n_in; (void)out_size; (void)d_ws; (void)ws_size;
  const float* value = (const float*)d_in[0];
  const float* query = (const float*)d_in[1];
  const float* Wq    = (const float*)d_in[2];
  const float* bq    = (const float*)d_in[3];
  const float* Wo    = (const float*)d_in[4];
  const float* bo    = (const float*)d_in[5];
  const int*   nn    = (const int*)d_in[6];
  float* out = (float*)d_out;
  graph_attn_kernel<<<dim3(NBATCH), dim3(256), 0, stream>>>(
      value, query, Wq, bq, Wo, bo, nn, out);
}
